// SelfAttention_50972671869136
// MI455X (gfx1250) — matrix operand order repaired and checked by reading, not compiled, not run on hardware
//
#include <hip/hip_runtime.h>
#include <hip/hip_bf16.h>

typedef _Float16 half_t;
typedef __attribute__((ext_vector_type(16))) _Float16 v16h;
typedef __attribute__((ext_vector_type(8)))  _Float16 v8h;
typedef __attribute__((ext_vector_type(8)))  float    v8f;

#define B_  4
#define S_  2048
#define E_  1024
#define H_  16
#define D_  64
#define F3_ (3 * E_)

// ---------------------------------------------------------------------------
// WMMA fragment helpers (wave32, v_wmma_f32_16x16x32_f16).
// A operand (16x32 f16, row-major, ld elems):
//   lane: row = lane&15, half = lane>>4
//   a[0..7]  = A[row][half*8 + 0..7]
//   a[8..15] = A[row][16 + half*8 + 0..7]
// B operand (32x16 f16) from BT = B^T row-major [N][K]:
//   lane: n = lane&15, khalf = lane>>4
//   b[0..7]  = BT[n][khalf*8 + 0..7]
//   b[8..15] = BT[n][16 + khalf*8 + 0..7]
// ---------------------------------------------------------------------------
__device__ __forceinline__ v16h frag_a(const half_t* __restrict__ base, int ld, int lane) {
  const int row = lane & 15, lh = lane >> 4;
  const half_t* p = base + row * ld + lh * 8;
  v8h lo = *(const v8h*)(p);
  v8h hi = *(const v8h*)(p + 16);
  v16h a;
#pragma unroll
  for (int i = 0; i < 8; ++i) { a[i] = lo[i]; a[i + 8] = hi[i]; }
  return a;
}

__device__ __forceinline__ v16h frag_b(const half_t* __restrict__ base, int ld, int lane) {
  const int row = lane & 15, khalf = lane >> 4;
  const half_t* p = base + row * ld + khalf * 8;
  v8h lo = *(const v8h*)(p);
  v8h hi = *(const v8h*)(p + 16);
  v16h b;
#pragma unroll
  for (int i = 0; i < 8; ++i) { b[i] = lo[i]; b[i + 8] = hi[i]; }
  return b;
}

__device__ __forceinline__ v8f wmma_f16(v16h a, v16h b, v8f c) {
  return __builtin_amdgcn_wmma_f32_16x16x32_f16(false, a, false, b, (short)0, c,
                                                false, false);
}

// ---------------------------------------------------------------------------
// fp32 -> f16 conversion / transpose
// ---------------------------------------------------------------------------
__global__ void cvt_f32_f16(const float* __restrict__ in, half_t* __restrict__ out, int n) {
  int i = blockIdx.x * blockDim.x + threadIdx.x;
  int stride = gridDim.x * blockDim.x;
  for (; i < n; i += stride) out[i] = (half_t)in[i];
}

// in: [rows][cols] fp32  ->  out: [cols][rows] f16
__global__ void transpose_cvt(const float* __restrict__ in, half_t* __restrict__ out,
                              int rows, int cols) {
  int i = blockIdx.x * blockDim.x + threadIdx.x;
  int stride = gridDim.x * blockDim.x;
  const int n = rows * cols;
  for (; i < n; i += stride) {
    int f = i / rows;
    int e = i - f * rows;
    out[i] = (half_t)in[(size_t)e * cols + f];
  }
}

// ---------------------------------------------------------------------------
// QKV projection: X[BS,E] @ Wqkv[E,3E] + b.  32x64 tile per wave:
// 8 independent accumulators, A frags reused 4x, B frags reused 2x.
// Wt is W^T f16 [3E][E].  Results scattered f16 into:
//   Q,K : [B][H][S][D]   (score-GEMM operand layout)
//   Vt  : [B][H][D][S]   (P@V B-operand layout)
// ---------------------------------------------------------------------------
__global__ __launch_bounds__(256) void qkv_gemm(
    const half_t* __restrict__ Xh, const half_t* __restrict__ Wt,
    const float* __restrict__ bias, half_t* __restrict__ Q,
    half_t* __restrict__ Km, half_t* __restrict__ Vt) {
  const int lane = threadIdx.x & 31;
  const int g = blockIdx.x * 8 + (threadIdx.x >> 5);
  const int NT = F3_ / 64;  // 48
  const int tm = g / NT, tn = g % NT;

  const half_t* abase0 = Xh + (size_t)(tm * 32) * E_;
  const half_t* abase1 = abase0 + (size_t)16 * E_;
  const half_t* bbase  = Wt + (size_t)(tn * 64) * E_;

  v8f acc[2][4];
#pragma unroll
  for (int mi = 0; mi < 2; ++mi)
#pragma unroll
    for (int t = 0; t < 4; ++t) acc[mi][t] = (v8f){};

#pragma unroll 2
  for (int k = 0; k < E_; k += 32) {
    v16h a0 = frag_a(abase0 + k, E_, lane);
    v16h a1 = frag_a(abase1 + k, E_, lane);
#pragma unroll
    for (int t = 0; t < 4; ++t) {
      v16h b = frag_b(bbase + (size_t)(t * 16) * E_ + k, E_, lane);
      acc[0][t] = wmma_f16(a0, b, acc[0][t]);
      acc[1][t] = wmma_f16(a1, b, acc[1][t]);
    }
  }

  const int n = lane & 15, lh = lane >> 4;
  // tn*64 block is 64-aligned -> same q/k/v segment and same head for all 4 tiles
  const int f0 = tn * 64;
  const int which = f0 >> 10;       // 0=Q 1=K 2=V
  const int h = (f0 & (E_ - 1)) >> 6;
#pragma unroll
  for (int t = 0; t < 4; ++t) {
    const int f = f0 + t * 16 + n;
    const float bv = bias[f];
    const int d = (f & 63);
#pragma unroll
    for (int mi = 0; mi < 2; ++mi) {
#pragma unroll
      for (int j = 0; j < 8; ++j) {
        const int gm = tm * 32 + mi * 16 + j + lh * 8;  // global token row
        const int bb = gm >> 11, s = gm & (S_ - 1);
        const int bh = bb * H_ + h;
        const half_t val = (half_t)(acc[mi][t][j] + bv);
        if (which == 0)      Q [((size_t)bh * S_ + s) * D_ + d] = val;
        else if (which == 1) Km[((size_t)bh * S_ + s) * D_ + d] = val;
        else                 Vt[((size_t)bh * D_ + d) * S_ + s] = val;
      }
    }
  }
}

// ---------------------------------------------------------------------------
// Flash attention: one wave per (b, h, 32-query tile).  S=2048, D=64.
// Per 32-key chunk: 8 score WMMAs + online softmax + 8 P@V WMMAs;
// K/V fragments shared across the two 16-query sub-tiles.
// P (2 x 16x32) goes through per-wave LDS tiles for C-layout -> A-layout.
// ---------------------------------------------------------------------------
__global__ __launch_bounds__(256) void attention(
    const half_t* __restrict__ Q, const half_t* __restrict__ Km,
    const half_t* __restrict__ Vt, half_t* __restrict__ O) {
  __shared__ _Float16 ldsP[8][2 * 16 * 32];
  const int lane = threadIdx.x & 31;
  const int wid = threadIdx.x >> 5;
  const int g = blockIdx.x * 8 + wid;
  const int QT = S_ / 32;  // 64
  const int bh = g / QT, qt = g % QT;

  const half_t* Qb = Q  + (size_t)bh * S_ * D_;
  const half_t* Kb = Km + (size_t)bh * S_ * D_;
  const half_t* Vb = Vt + (size_t)bh * D_ * S_;
  _Float16* P0 = ldsP[wid];
  _Float16* P1 = ldsP[wid] + 16 * 32;

  // Q fragments: [mi][d-chunk]
  v16h aQ[2][2];
#pragma unroll
  for (int mi = 0; mi < 2; ++mi)
#pragma unroll
    for (int d2 = 0; d2 < 2; ++d2)
      aQ[mi][d2] = frag_a(Qb + (size_t)(qt * 32 + mi * 16) * D_ + d2 * 32, D_, lane);

  v8f acc[2][4];
#pragma unroll
  for (int mi = 0; mi < 2; ++mi)
#pragma unroll
    for (int t = 0; t < 4; ++t) acc[mi][t] = (v8f){};

  float mrow[2][8], lrow[2][8];
#pragma unroll
  for (int mi = 0; mi < 2; ++mi)
#pragma unroll
    for (int j = 0; j < 8; ++j) { mrow[mi][j] = -1e30f; lrow[mi][j] = 0.f; }

  const int n = lane & 15, lh = lane >> 4;
  const float scale = 0.125f;  // 1/sqrt(64)

  for (int kb = 0; kb < S_; kb += 32) {
    // --- scores: 32 queries x 32 keys -------------------------------------
    v8f s[2][2];
#pragma unroll
    for (int mi = 0; mi < 2; ++mi)
#pragma unroll
      for (int n2 = 0; n2 < 2; ++n2) s[mi][n2] = (v8f){};
#pragma unroll
    for (int n2 = 0; n2 < 2; ++n2) {
#pragma unroll
      for (int d2 = 0; d2 < 2; ++d2) {
        v16h bK = frag_b(Kb + (size_t)(kb + n2 * 16) * D_ + d2 * 32, D_, lane);
        s[0][n2] = wmma_f16(aQ[0][d2], bK, s[0][n2]);
        s[1][n2] = wmma_f16(aQ[1][d2], bK, s[1][n2]);
      }
    }

    // --- online softmax (rows split by lane-half, cols across 16 lanes) ---
#pragma unroll
    for (int mi = 0; mi < 2; ++mi) {
      _Float16* P = mi ? P1 : P0;
#pragma unroll
      for (int j = 0; j < 8; ++j) {
        float x0 = s[mi][0][j] * scale, x1 = s[mi][1][j] * scale;
        float mx = fmaxf(x0, x1);
        mx = fmaxf(mx, __shfl_xor(mx, 1));
        mx = fmaxf(mx, __shfl_xor(mx, 2));
        mx = fmaxf(mx, __shfl_xor(mx, 4));
        mx = fmaxf(mx, __shfl_xor(mx, 8));
        const float mnew = fmaxf(mrow[mi][j], mx);
        const float alpha = __expf(mrow[mi][j] - mnew);
        const float p0 = __expf(x0 - mnew);
        const float p1 = __expf(x1 - mnew);
        float rs = p0 + p1;
        rs += __shfl_xor(rs, 1);
        rs += __shfl_xor(rs, 2);
        rs += __shfl_xor(rs, 4);
        rs += __shfl_xor(rs, 8);
        lrow[mi][j] = lrow[mi][j] * alpha + rs;
        mrow[mi][j] = mnew;
        acc[mi][0][j] *= alpha; acc[mi][1][j] *= alpha;
        acc[mi][2][j] *= alpha; acc[mi][3][j] *= alpha;
        const int m = j + lh * 8;
        P[m * 32 + n]      = (half_t)p0;
        P[m * 32 + 16 + n] = (half_t)p1;
      }
    }
    asm volatile("s_wait_dscnt 0" ::: "memory");

    // --- reload P tiles as A-operand fragments ----------------------------
    v16h aP[2];
#pragma unroll
    for (int mi = 0; mi < 2; ++mi) {
      const _Float16* pp = (mi ? P1 : P0) + (lane & 15) * 32 + lh * 8;
      v8h lo = *(const v8h*)(pp);
      v8h hi = *(const v8h*)(pp + 16);
#pragma unroll
      for (int i = 0; i < 8; ++i) { aP[mi][i] = lo[i]; aP[mi][i + 8] = hi[i]; }
    }

    // --- O(32x64) += P(32x32) @ V(32x64); Vt rows are d, contiguous keys ---
#pragma unroll
    for (int t = 0; t < 4; ++t) {
      v16h bV = frag_b(Vb + (size_t)(t * 16) * S_ + kb, S_, lane);
      acc[0][t] = wmma_f16(aP[0], bV, acc[0][t]);
      acc[1][t] = wmma_f16(aP[1], bV, acc[1][t]);
    }
    asm volatile("s_wait_dscnt 0" ::: "memory");
  }

  // --- normalize, store O as f16 [B*S][E] (A-operand-ready for out proj) ---
  const int bb = bh / H_, h = bh % H_;
#pragma unroll
  for (int mi = 0; mi < 2; ++mi) {
#pragma unroll
    for (int j = 0; j < 8; ++j) {
      const int s = qt * 32 + mi * 16 + j + lh * 8;
      const float inv = 1.f / lrow[mi][j];
      const size_t base = ((size_t)(bb * S_ + s)) * E_ + h * D_;
      O[base + 0 * 16 + n] = (half_t)(acc[mi][0][j] * inv);
      O[base + 1 * 16 + n] = (half_t)(acc[mi][1][j] * inv);
      O[base + 2 * 16 + n] = (half_t)(acc[mi][2][j] * inv);
      O[base + 3 * 16 + n] = (half_t)(acc[mi][3][j] * inv);
    }
  }
}

// ---------------------------------------------------------------------------
// Output projection: O[BS,E] @ Wout[E,E] + b -> fp32 out.  32x64 per wave.
// ---------------------------------------------------------------------------
__global__ __launch_bounds__(256) void out_gemm(
    const half_t* __restrict__ O, const half_t* __restrict__ Wt,
    const float* __restrict__ bias, float* __restrict__ out) {
  const int lane = threadIdx.x & 31;
  const int g = blockIdx.x * 8 + (threadIdx.x >> 5);
  const int NT = E_ / 64;  // 16
  const int tm = g / NT, tn = g % NT;

  const half_t* abase0 = O + (size_t)(tm * 32) * E_;
  const half_t* abase1 = abase0 + (size_t)16 * E_;
  const half_t* bbase  = Wt + (size_t)(tn * 64) * E_;

  v8f acc[2][4];
#pragma unroll
  for (int mi = 0; mi < 2; ++mi)
#pragma unroll
    for (int t = 0; t < 4; ++t) acc[mi][t] = (v8f){};

#pragma unroll 2
  for (int k = 0; k < E_; k += 32) {
    v16h a0 = frag_a(abase0 + k, E_, lane);
    v16h a1 = frag_a(abase1 + k, E_, lane);
#pragma unroll
    for (int t = 0; t < 4; ++t) {
      v16h b = frag_b(bbase + (size_t)(t * 16) * E_ + k, E_, lane);
      acc[0][t] = wmma_f16(a0, b, acc[0][t]);
      acc[1][t] = wmma_f16(a1, b, acc[1][t]);
    }
  }

  const int n = lane & 15, lh = lane >> 4;
#pragma unroll
  for (int t = 0; t < 4; ++t) {
    const int f = tn * 64 + t * 16 + n;
    const float bv = bias[f];
#pragma unroll
    for (int mi = 0; mi < 2; ++mi) {
#pragma unroll
      for (int j = 0; j < 8; ++j) {
        const int gm = tm * 32 + mi * 16 + j + lh * 8;
        out[(size_t)gm * E_ + f] = acc[mi][t][j] + bv;
      }
    }
  }
}

// ---------------------------------------------------------------------------
extern "C" void kernel_launch(void* const* d_in, const int* in_sizes, int n_in,
                              void* d_out, int out_size, void* d_ws, size_t ws_size,
                              hipStream_t stream) {
  const float* x    = (const float*)d_in[0];  // [B,S,E]
  const float* wqkv = (const float*)d_in[1];  // [E,3E]
  const float* bqkv = (const float*)d_in[2];  // [3E]
  const float* wout = (const float*)d_in[3];  // [E,E]
  const float* bout = (const float*)d_in[4];  // [E]

  char* ws = (char*)d_ws;
  size_t off = 0;
  half_t* Xh  = (half_t*)(ws + off); off += (size_t)B_ * S_ * E_ * 2;
  half_t* Wqt = (half_t*)(ws + off); off += (size_t)F3_ * E_ * 2;
  half_t* Wot = (half_t*)(ws + off); off += (size_t)E_ * E_ * 2;
  half_t* Qm  = (half_t*)(ws + off); off += (size_t)B_ * H_ * S_ * D_ * 2;
  half_t* Km  = (half_t*)(ws + off); off += (size_t)B_ * H_ * S_ * D_ * 2;
  half_t* Vtm = (half_t*)(ws + off); off += (size_t)B_ * H_ * D_ * S_ * 2;
  half_t* Om  = (half_t*)(ws + off); off += (size_t)B_ * S_ * E_ * 2;
  (void)ws_size; (void)in_sizes; (void)n_in; (void)out_size;

  cvt_f32_f16  <<<4096, 256, 0, stream>>>(x, Xh, B_ * S_ * E_);
  transpose_cvt<<<4096, 256, 0, stream>>>(wqkv, Wqt, E_, F3_);
  transpose_cvt<<<2048, 256, 0, stream>>>(wout, Wot, E_, E_);

  const int qkv_waves = ((B_ * S_) / 32) * (F3_ / 64);      // 256*48 = 12288
  qkv_gemm<<<qkv_waves / 8, 256, 0, stream>>>(Xh, Wqt, bqkv, Qm, Km, Vtm);

  const int attn_waves = B_ * H_ * (S_ / 32);               // 4096
  attention<<<attn_waves / 8, 256, 0, stream>>>(Qm, Km, Vtm, Om);

  const int out_waves = ((B_ * S_) / 32) * (E_ / 64);       // 256*16 = 4096
  out_gemm<<<out_waves / 8, 256, 0, stream>>>(Om, Wot, bout, (float*)d_out);
}
